// ParallelBlock_27805618275138
// MI455X (gfx1250) — matrix operand order repaired and checked by reading, not compiled, not run on hardware
//
#include <hip/hip_runtime.h>
#include <hip/hip_bf16.h>
#include <math.h>

typedef __bf16 bf16;
typedef __bf16 v16bf __attribute__((ext_vector_type(16)));
typedef float  v8f   __attribute__((ext_vector_type(8)));
typedef int    v4i   __attribute__((ext_vector_type(4)));

typedef __attribute__((address_space(1))) void as1_void;
typedef __attribute__((address_space(3))) void as3_void;
typedef __attribute__((address_space(1))) v4i  as1_v4i;
typedef __attribute__((address_space(3))) v4i  as3_v4i;

#define DEVINL __device__ __forceinline__

#if defined(__AMDGCN__) && __has_builtin(__builtin_amdgcn_global_load_async_to_lds_b128) && \
    __has_builtin(__builtin_amdgcn_s_wait_asynccnt)
#define USE_ASYNC_LDS 1
#else
#define USE_ASYNC_LDS 0
#endif

// ---------------- problem constants ----------------
static constexpr int Cdim = 512;
static constexpr int NT2 = 1 + 48 * 48;   // 2305
static constexpr int NT3 = 1 + 24 * 24;   // 577
static constexpr int NT4 = 1 + 12 * 12;   // 145
static constexpr int NT1 = 1 + 96 * 96;   // 9217
static constexpr int Mrows2 = 16 * NT2;
static constexpr int Mrows3 = 16 * NT3;
static constexpr int Mrows4 = 16 * NT4;

// ---------------- reductions ----------------
DEVINL float blockReduceSum256(float v, float* red) {
    int tid = threadIdx.x;
    #pragma unroll
    for (int off = 16; off > 0; off >>= 1) v += __shfl_down(v, off, 32);
    if ((tid & 31) == 0) red[tid >> 5] = v;
    __syncthreads();
    float t = (tid < 8) ? red[tid] : 0.0f;
    #pragma unroll
    for (int off = 4; off > 0; off >>= 1) t += __shfl_down(t, off, 32);
    if (tid == 0) red[0] = t;
    __syncthreads();
    float r = red[0];
    __syncthreads();
    return r;
}

DEVINL float blockReduceMax256(float v, float* red) {
    int tid = threadIdx.x;
    #pragma unroll
    for (int off = 16; off > 0; off >>= 1) v = fmaxf(v, __shfl_down(v, off, 32));
    if ((tid & 31) == 0) red[tid >> 5] = v;
    __syncthreads();
    float t = (tid < 8) ? red[tid] : -INFINITY;
    #pragma unroll
    for (int off = 4; off > 0; off >>= 1) t = fmaxf(t, __shfl_down(t, off, 32));
    if (tid == 0) red[0] = t;
    __syncthreads();
    float r = red[0];
    __syncthreads();
    return r;
}

// ---------------- fp32 -> bf16 convert ----------------
__global__ __launch_bounds__(256) void cvt_f32_bf16(const float* __restrict__ in,
                                                    bf16* __restrict__ out, size_t n) {
    size_t i = (size_t)blockIdx.x * 256 + threadIdx.x;
    if (i < n) out[i] = (bf16)in[i];
}

// ---------------- LayerNorm (C=512) -> bf16 ----------------
__global__ __launch_bounds__(256) void layernorm_bf16(const float* __restrict__ x,
                                                      const float* __restrict__ g,
                                                      const float* __restrict__ b,
                                                      bf16* __restrict__ out) {
    __shared__ float red[8];
    const int row = blockIdx.x;
    const int tid = threadIdx.x;
    const float* xr = x + (size_t)row * Cdim;
    float a0 = xr[tid], a1 = xr[tid + 256];
    float mean = blockReduceSum256(a0 + a1, red) * (1.0f / 512.0f);
    float d0 = a0 - mean, d1 = a1 - mean;
    float var = blockReduceSum256(d0 * d0 + d1 * d1, red) * (1.0f / 512.0f);
    float rstd = rsqrtf(var + 1e-5f);
    bf16* o = out + (size_t)row * Cdim;
    o[tid]       = (bf16)(d0 * rstd * g[tid]       + b[tid]);
    o[tid + 256] = (bf16)(d1 * rstd * g[tid + 256] + b[tid + 256]);
}

// ---------------- WMMA bf16 GEMM ----------------
// out = act(A[M,K] @ W[N,K]^T + bias) (+res). A,W bf16 row-major. N%128==0, K%32==0.
template <bool GELU>
__global__ __launch_bounds__(256) void gemm_bf16_wmma(const bf16* __restrict__ A,
                                                      const bf16* __restrict__ W,
                                                      const float* __restrict__ bias,
                                                      const float* __restrict__ res,
                                                      float* __restrict__ outF,
                                                      bf16* __restrict__ outB,
                                                      int M, int N, int K) {
    constexpr int BM = 64, BN = 128, BK = 32, LDA = 40, LDB = 40;
    __shared__ bf16 sA[BM * LDA];
    __shared__ bf16 sB[BN * LDB];

    const int tid  = threadIdx.x;
    const int lane = tid & 31;
    const int wave = tid >> 5;
    const int wr   = wave & 1;   // wave row (0..1)  -> 32 rows each
    const int wc   = wave >> 1;  // wave col (0..3)  -> 32 cols each
    const int m0   = blockIdx.x * BM;
    const int n0   = blockIdx.y * BN;
    const int lm   = lane & 15;
    const int half = lane >> 4;

    const int aRow = tid >> 2;        // 0..63
    const int aCol = (tid & 3) * 8;   // 0,8,16,24

    v8f acc[2][2] = {};

    union Frag { unsigned int u[8]; v16bf v; };
    auto computeStep = [&]() {
        #pragma unroll
        for (int fi = 0; fi < 2; ++fi) {
            Frag af;
            const bf16* ab = sA + (wr * 32 + fi * 16 + lm) * LDA;
            #pragma unroll
            for (int v = 0; v < 8; ++v) {
                const int k0 = (v < 4) ? (2 * v + 8 * half) : (16 + 2 * (v - 4) + 8 * half);
                af.u[v] = *(const unsigned int*)(ab + k0);
            }
            #pragma unroll
            for (int fj = 0; fj < 2; ++fj) {
                Frag bfq;
                const bf16* bb = sB + (wc * 32 + fj * 16 + lm) * LDB;
                #pragma unroll
                for (int v = 0; v < 8; ++v) {
                    const int k0 = (v < 4) ? (2 * v + 8 * half) : (16 + 2 * (v - 4) + 8 * half);
                    bfq.u[v] = *(const unsigned int*)(bb + k0);
                }
                acc[fi][fj] = __builtin_amdgcn_wmma_f32_16x16x32_bf16(
                    false, af.v, false, bfq.v, (short)0, acc[fi][fj], false, false);
            }
        }
    };

    const bool fullTile = (m0 + BM <= M);

#if USE_ASYNC_LDS
    if (fullTile) {
        // gfx1250 async global->LDS path: one ASYNC op replaces load+ds_store,
        // tracked with ASYNCcnt (s_wait_asynccnt), no VGPR staging.
        for (int kb = 0; kb < K; kb += BK) {
            __syncthreads();
            __builtin_amdgcn_global_load_async_to_lds_b128(
                (as1_v4i*)(as1_void*)(A + (size_t)(m0 + aRow) * K + kb + aCol),
                (as3_v4i*)(as3_void*)(sA + aRow * LDA + aCol), 0, 0);
            #pragma unroll
            for (int i = 0; i < 2; ++i) {
                const int ch  = tid + i * 256;
                const int r   = ch >> 2;
                const int col = (ch & 3) * 8;
                __builtin_amdgcn_global_load_async_to_lds_b128(
                    (as1_v4i*)(as1_void*)(W + (size_t)(n0 + r) * K + kb + col),
                    (as3_v4i*)(as3_void*)(sB + r * LDB + col), 0, 0);
            }
            __builtin_amdgcn_s_wait_asynccnt(0);
            __syncthreads();
            computeStep();
        }
    } else
#endif
    {
        // fallback / ragged-M path (zero-fills OOB rows in LDS)
        for (int kb = 0; kb < K; kb += BK) {
            __syncthreads();
            {
                const int gr = m0 + aRow;
                uint4 val = {0u, 0u, 0u, 0u};
                if (gr < M) val = *(const uint4*)(A + (size_t)gr * K + kb + aCol);
                *(uint4*)(sA + aRow * LDA + aCol) = val;
            }
            #pragma unroll
            for (int i = 0; i < 2; ++i) {
                const int ch  = tid + i * 256;
                const int r   = ch >> 2;
                const int col = (ch & 3) * 8;
                uint4 val = *(const uint4*)(W + (size_t)(n0 + r) * K + kb + col);
                *(uint4*)(sB + r * LDB + col) = val;
            }
            if (kb + 2 * BK < K) {  // deepen pipeline: L2 prefetch of K+2 tile
                __builtin_prefetch(W + (size_t)(n0 + (tid >> 2)) * K + kb + 2 * BK, 0, 1);
            }
            __syncthreads();
            computeStep();
        }
    }

    // ---- epilogue ----
    #pragma unroll
    for (int fi = 0; fi < 2; ++fi) {
        #pragma unroll
        for (int fj = 0; fj < 2; ++fj) {
            const int col = n0 + wc * 32 + fj * 16 + lm;
            #pragma unroll
            for (int v = 0; v < 8; ++v) {
                const int row = m0 + wr * 32 + fi * 16 + v + 8 * half;
                if (row < M) {
                    float val = acc[fi][fj][v];
                    if (bias) val += bias[col];
                    if (GELU) val = 0.5f * val * (1.0f + erff(val * 0.70710678118f));
                    const size_t idx = (size_t)row * N + col;
                    if (res)  val += res[idx];
                    if (outF) outF[idx] = val;
                    if (outB) outB[idx] = (bf16)val;
                }
            }
        }
    }
}

// ---------------- column softmax stats over K part of qkv ----------------
// column id = b*512 + h*64 + d ; values qkv[b*Ntok+n][512 + h*64+d]
__global__ __launch_bounds__(256) void ksoftmax_stats(const float* __restrict__ qkv,
                                                      float* __restrict__ stats, int Ntok) {
    __shared__ float red[8];
    const int col = blockIdx.x;
    const int b = col >> 9, c = col & 511;
    const float* base = qkv + (size_t)b * Ntok * 1536 + 512 + c;
    const int tid = threadIdx.x;
    float mx = -INFINITY;
    for (int n = tid; n < Ntok; n += 256) mx = fmaxf(mx, base[(size_t)n * 1536]);
    mx = blockReduceMax256(mx, red);
    float s = 0.0f;
    for (int n = tid; n < Ntok; n += 256) s += expf(base[(size_t)n * 1536] - mx);
    s = blockReduceSum256(s, red);
    if (tid == 0) { stats[col * 2] = mx; stats[col * 2 + 1] = s; }
}

// ---------------- fa[b,h,ck,d] = sum_n softmax(k)[n,ck] * v[n,d] ----------------
__global__ __launch_bounds__(256) void fa_kernel(const float* __restrict__ qkv,
                                                 const float* __restrict__ stats,
                                                 float* __restrict__ fa, int Ntok) {
    __shared__ float sk[32 * 64];
    __shared__ float sv[32 * 64];
    const int bh = blockIdx.x;            // b*8 + h
    const int b = bh >> 3, h = bh & 7;
    const float* kbase = qkv + (size_t)b * Ntok * 1536 + 512  + h * 64;
    const float* vbase = qkv + (size_t)b * Ntok * 1536 + 1024 + h * 64;
    const float* st = stats + (size_t)(b * 512 + h * 64) * 2;
    const int tid = threadIdx.x;
    const int ck  = tid >> 2;             // 0..63
    const int cd0 = (tid & 3) * 16;
    const float mx  = st[ck * 2];
    const float inv = 1.0f / st[ck * 2 + 1];
    float accv[16];
    #pragma unroll
    for (int j = 0; j < 16; ++j) accv[j] = 0.0f;

    for (int n0 = 0; n0 < Ntok; n0 += 32) {
        const int cnt = min(32, Ntok - n0);
        __syncthreads();
        for (int i = tid; i < 32 * 64; i += 256) {
            const int nn = i >> 6, dd = i & 63;
            float kv = 0.0f, vv = 0.0f;
            if (nn < cnt) {
                const size_t roff = (size_t)(n0 + nn) * 1536;
                kv = kbase[roff + dd];
                vv = vbase[roff + dd];
            }
            sk[i] = kv; sv[i] = vv;
        }
        __syncthreads();
        for (int nn = 0; nn < cnt; ++nn) {
            const float ks = expf(sk[nn * 64 + ck] - mx) * inv;
            const float* vr = sv + nn * 64 + cd0;
            #pragma unroll
            for (int j = 0; j < 16; ++j) accv[j] += ks * vr[j];
        }
    }
    float* o = fa + ((size_t)bh * 64 + ck) * 64 + cd0;
    #pragma unroll
    for (int j = 0; j < 16; ++j) o[j] = accv[j];
}

// ---------------- attn = 0.125*(q@fa) + crpe(q,v) -> bf16 ----------------
__global__ __launch_bounds__(256) void attn_crpe_kernel(const float* __restrict__ qkv,
                                                        const float* __restrict__ fa,
                                                        const float* __restrict__ cw0,
                                                        const float* __restrict__ cw1,
                                                        const float* __restrict__ cw2,
                                                        const float* __restrict__ cb0,
                                                        const float* __restrict__ cb1,
                                                        const float* __restrict__ cb2,
                                                        bf16* __restrict__ out,
                                                        int Ntok, int H, int W) {
    __shared__ float sq[512];
    const int m = blockIdx.x;                 // b*Ntok + n
    const int b = m / Ntok, n = m - b * Ntok;
    const int tid = threadIdx.x;
    const float* qrow = qkv + (size_t)m * 1536;
    sq[tid] = qrow[tid];
    sq[tid + 256] = qrow[tid + 256];
    __syncthreads();

    for (int c = tid; c < 512; c += 256) {
        const int h = c >> 6, d = c & 63;
        const float* farow = fa + ((size_t)(b * 8 + h) * 64) * 64 + d;  // stride 64 over k
        const float* qh = sq + h * 64;
        float accq = 0.0f;
        #pragma unroll 8
        for (int k = 0; k < 64; ++k) accq += qh[k] * farow[(size_t)k * 64];
        float val = 0.125f * accq;

        if (n > 0) {
            const int idx = n - 1;
            const int y = idx / W, x = idx - y * W;
            int ks; const float* wt; float bconv;
            if (c < 128)      { ks = 3; wt = cw0 + c * 9;          bconv = cb0[c]; }
            else if (c < 320) { ks = 5; wt = cw1 + (c - 128) * 25; bconv = cb1[c - 128]; }
            else              { ks = 7; wt = cw2 + (c - 320) * 49; bconv = cb2[c - 320]; }
            const int pad = ks >> 1;
            const float* vimg = qkv + ((size_t)b * Ntok + 1) * 1536 + 1024 + c;
            float conv = 0.0f;
            for (int ky = 0; ky < ks; ++ky) {
                const int yy = y + ky - pad;
                if (yy < 0 || yy >= H) continue;
                for (int kx = 0; kx < ks; ++kx) {
                    const int xx = x + kx - pad;
                    if (xx < 0 || xx >= W) continue;
                    conv += wt[ky * ks + kx] * vimg[(size_t)(yy * W + xx) * 1536];
                }
            }
            conv += bconv;
            val += sq[c] * conv;
        }
        out[(size_t)m * 512 + c] = (bf16)val;
    }
}

// ---------------- out = a + b ----------------
__global__ __launch_bounds__(256) void add2_kernel(const float* __restrict__ a,
                                                   const float* __restrict__ b,
                                                   float* __restrict__ o, size_t n) {
    size_t i = (size_t)blockIdx.x * 256 + threadIdx.x;
    if (i < n) o[i] = a[i] + b[i];
}

// ---------------- dst += bilinear_resize(src) (half-pixel, edge clamp) ----------------
__global__ __launch_bounds__(256) void add_bilinear(float* __restrict__ dst,
                                                    const float* __restrict__ src,
                                                    int sH, int sW, int dH, int dW) {
    const int dN = 1 + dH * dW;
    const int sN = 1 + sH * sW;
    const size_t total = (size_t)16 * dN * 512;
    size_t i = (size_t)blockIdx.x * 256 + threadIdx.x;
    if (i >= total) return;
    const int c = (int)(i & 511);
    const size_t t = i >> 9;
    const int b = (int)(t / dN);
    const int n = (int)(t - (size_t)b * dN);
    float v;
    if (n == 0) {
        v = src[(size_t)b * sN * 512 + c];
    } else {
        const int idx = n - 1;
        const int y = idx / dW, x = idx - y * dW;
        float fy = (y + 0.5f) * (float)sH / (float)dH - 0.5f;
        float fx = (x + 0.5f) * (float)sW / (float)dW - 0.5f;
        fy = fminf(fmaxf(fy, 0.0f), (float)(sH - 1));
        fx = fminf(fmaxf(fx, 0.0f), (float)(sW - 1));
        const int y0 = (int)floorf(fy), x0 = (int)floorf(fx);
        const int y1 = min(y0 + 1, sH - 1), x1 = min(x0 + 1, sW - 1);
        const float wy = fy - y0, wx = fx - x0;
        const float* sb = src + (size_t)b * sN * 512;
        const float v00 = sb[(size_t)(1 + y0 * sW + x0) * 512 + c];
        const float v01 = sb[(size_t)(1 + y0 * sW + x1) * 512 + c];
        const float v10 = sb[(size_t)(1 + y1 * sW + x0) * 512 + c];
        const float v11 = sb[(size_t)(1 + y1 * sW + x1) * 512 + c];
        v = (1.0f - wy) * ((1.0f - wx) * v00 + wx * v01) + wy * ((1.0f - wx) * v10 + wx * v11);
    }
    dst[i] += v;
}

// =====================================================================
extern "C" void kernel_launch(void* const* d_in, const int* in_sizes, int n_in,
                              void* d_out, int out_size, void* d_ws, size_t ws_size,
                              hipStream_t stream) {
    (void)in_sizes; (void)n_in; (void)out_size; (void)ws_size;

    // ---- input indices ----
    const float* x_in[3] = { (const float*)d_in[1], (const float*)d_in[2], (const float*)d_in[3] };
    const int    pbase[3] = { 4, 17, 30 };        // p2, p3, p4 (13 arrays each)
    const int    Ntoks[3] = { NT2, NT3, NT4 };
    const int    Hs[3]    = { 48, 24, 12 };
    const int    Mr[3]    = { Mrows2, Mrows3, Mrows4 };
    const float* fc1_w = (const float*)d_in[43];
    const float* fc1_b = (const float*)d_in[44];
    const float* fc2_w = (const float*)d_in[45];
    const float* fc2_b = (const float*)d_in[46];

    // ---- workspace carve-up ----
    char* wp = (char*)d_ws;
    auto alloc = [&](size_t bytes) -> void* {
        void* r = (void*)wp;
        wp += (bytes + 255) & ~(size_t)255;
        return r;
    };
    bf16* wqkv[3]; bf16* wproj[3];
    for (int i = 0; i < 3; ++i) {
        wqkv[i]  = (bf16*)alloc((size_t)1536 * 512 * 2);
        wproj[i] = (bf16*)alloc((size_t)512 * 512 * 2);
    }
    bf16* wfc1 = (bf16*)alloc((size_t)2048 * 512 * 2);
    bf16* wfc2 = (bf16*)alloc((size_t)512 * 2048 * 2);
    float* cur[3];
    for (int i = 0; i < 3; ++i) cur[i] = (float*)alloc((size_t)Mr[i] * 512 * 4);
    bf16*  xn      = (bf16*)alloc((size_t)Mrows2 * 512 * 2);
    float* qkvbuf  = (float*)alloc((size_t)Mrows2 * 1536 * 4);
    bf16*  attnbuf = (bf16*)alloc((size_t)Mrows2 * 512 * 2);
    bf16*  hbuf    = (bf16*)alloc((size_t)Mrows2 * 2048 * 2);
    float* stats   = (float*)alloc((size_t)16 * 512 * 2 * 4);
    float* fabuf   = (float*)alloc((size_t)16 * 8 * 64 * 64 * 4);

    auto cvt = [&](const float* src, bf16* dst, size_t n) {
        cvt_f32_bf16<<<dim3((unsigned)((n + 255) / 256)), 256, 0, stream>>>(src, dst, n);
    };
    // ---- weight conversion ----
    for (int i = 0; i < 3; ++i) {
        cvt((const float*)d_in[pbase[i] + 2], wqkv[i],  (size_t)1536 * 512);
        cvt((const float*)d_in[pbase[i] + 3], wproj[i], (size_t)512 * 512);
    }
    cvt(fc1_w, wfc1, (size_t)2048 * 512);
    cvt(fc2_w, wfc2, (size_t)512 * 2048);

    // ---- per-branch attention ----
    for (int i = 0; i < 3; ++i) {
        const int pb = pbase[i], M = Mr[i], Ntok = Ntoks[i], HW = Hs[i];
        const float* n1g = (const float*)d_in[pb + 0];
        const float* n1b = (const float*)d_in[pb + 1];
        const float* projb = (const float*)d_in[pb + 4];
        layernorm_bf16<<<M, 256, 0, stream>>>(x_in[i], n1g, n1b, xn);
        gemm_bf16_wmma<false><<<dim3((M + 63) / 64, 1536 / 128), 256, 0, stream>>>(
            xn, wqkv[i], nullptr, nullptr, qkvbuf, nullptr, M, 1536, 512);
        ksoftmax_stats<<<16 * 512, 256, 0, stream>>>(qkvbuf, stats, Ntok);
        fa_kernel<<<16 * 8, 256, 0, stream>>>(qkvbuf, stats, fabuf, Ntok);
        attn_crpe_kernel<<<M, 256, 0, stream>>>(
            qkvbuf, fabuf,
            (const float*)d_in[pb + 5], (const float*)d_in[pb + 6], (const float*)d_in[pb + 7],
            (const float*)d_in[pb + 8], (const float*)d_in[pb + 9], (const float*)d_in[pb + 10],
            attnbuf, Ntok, HW, HW);
        gemm_bf16_wmma<false><<<dim3((M + 63) / 64, 512 / 128), 256, 0, stream>>>(
            attnbuf, wproj[i], projb, nullptr, cur[i], nullptr, M, 512, 512);
    }

    // ---- output regions ----
    float* out1 = (float*)d_out;
    float* ox[3];
    ox[0] = out1 + (size_t)16 * NT1 * 512;
    ox[1] = ox[0] + (size_t)Mrows2 * 512;
    ox[2] = ox[1] + (size_t)Mrows3 * 512;

    // x1 passthrough
    (void)hipMemcpyAsync(out1, d_in[0], (size_t)16 * NT1 * 512 * 4,
                         hipMemcpyDeviceToDevice, stream);

    // ---- cross-scale fusion: ox[i] = x_i + cur_i + resized others ----
    for (int i = 0; i < 3; ++i) {
        const size_t n = (size_t)Mr[i] * 512;
        add2_kernel<<<dim3((unsigned)((n + 255) / 256)), 256, 0, stream>>>(x_in[i], cur[i], ox[i], n);
    }
    auto addbl = [&](float* dst, const float* src, int sH, int dH) {
        const size_t n = (size_t)16 * (1 + dH * dH) * 512;
        add_bilinear<<<dim3((unsigned)((n + 255) / 256)), 256, 0, stream>>>(dst, src, sH, sH, dH, dH);
    };
    addbl(ox[0], cur[1], 24, 48);  // c2 += up(cur3)
    addbl(ox[0], cur[2], 12, 48);  // c2 += up(cur4)
    addbl(ox[1], cur[2], 12, 24);  // c3 += up(cur4)
    addbl(ox[1], cur[0], 48, 24);  // c3 += down(cur2)
    addbl(ox[2], cur[1], 24, 12);  // c4 += down(cur3)
    addbl(ox[2], cur[0], 48, 12);  // c4 += down(cur2)

    // ---- MLP per branch (in-place residual on d_out regions) ----
    for (int i = 0; i < 3; ++i) {
        const int pb = pbase[i], M = Mr[i];
        const float* n2g = (const float*)d_in[pb + 11];
        const float* n2b = (const float*)d_in[pb + 12];
        layernorm_bf16<<<M, 256, 0, stream>>>(ox[i], n2g, n2b, xn);
        gemm_bf16_wmma<true><<<dim3((M + 63) / 64, 2048 / 128), 256, 0, stream>>>(
            xn, wfc1, fc1_b, nullptr, nullptr, hbuf, M, 2048, 512);
        gemm_bf16_wmma<false><<<dim3((M + 63) / 64, 512 / 128), 256, 0, stream>>>(
            hbuf, wfc2, fc2_b, ox[i], ox[i], nullptr, M, 512, 2048);
    }
}
